// MambaBlock_40982577938976
// MI455X (gfx1250) — matrix operand order repaired and checked by reading, not compiled, not run on hardware
//
#include <hip/hip_runtime.h>
#include <hip/hip_bf16.h>

// ---------------- shapes ----------------
#define B_      32
#define LSEQ    1024
#define DMODEL  64
#define DIN     256          // EXPAND*DMODEL
#define DSTATE  16
#define DCONV   4
#define NHEADS  4
#define HEADDIM 64
#define CONVDIM 288          // DIN + 2*DSTATE
#define DINPROJ 548          // 2*DIN + 2*DSTATE + NHEADS
#define NPAD1   560          // 35 tiles of 16
#define ROWS    (B_*LSEQ)    // 32768
#define NCHUNK  32           // scan chunks
#define CHS     32           // steps per chunk (NCHUNK*CHS == LSEQ)
#define EPS_    1e-5f

typedef _Float16 v8h  __attribute__((ext_vector_type(8)));
typedef _Float16 v16h __attribute__((ext_vector_type(16)));
typedef float    v8f  __attribute__((ext_vector_type(8)));

union CF { v8f v; float f[8]; };

// ---- WMMA fragment loaders (16x16x32 f16, wave32, contiguous K=16h+e order used for both A and B) ----
// A (16x32 MxK): lane l -> row m=l&15, half h=l>>4; elems e -> K=16h+e
// (elems 0..7 = K=16h..16h+7, elems 8..15 = K=16h+8..16h+15) -> two contiguous 16B loads.
__device__ __forceinline__ v16h load_a_frag(const _Float16* rowbase, int kk, int half) {
    union { v16h v; v8h h[2]; } a;
    a.h[0] = *(const v8h*)(rowbase + kk + 16 * half);
    a.h[1] = *(const v8h*)(rowbase + kk + 16 * half + 8);
    return a.v;
}
// B (32x16 KxN): lane l -> col n=l&15, half h=l>>4; elems e -> K=16h+e
// -> one contiguous 32B load along K (weights stored row-major per output col).
__device__ __forceinline__ v16h load_b_frag(const _Float16* colbase, int kk, int half) {
    return *(const v16h*)(colbase + kk + 16 * half);
}
__device__ __forceinline__ v8f wmma_f16(v16h a, v16h b, v8f c) {
    return __builtin_amdgcn_wmma_f32_16x16x32_f16(false, a, false, b,
                                                  (short)0, c, false, false);
}

// ---------------- kernel 1: weight prep (fp32 -> f16, pad in_proj to 560) ---
__global__ void k_prep_weights(const float* __restrict__ w_in,   // (548,64)
                               const float* __restrict__ w_out,  // (64,256)
                               _Float16* __restrict__ w1h,       // (560,64)
                               _Float16* __restrict__ w2h) {     // (64,256)
    int i = blockIdx.x * blockDim.x + threadIdx.x;
    const int n1 = NPAD1 * DMODEL;          // 35840
    const int n2 = DMODEL * DIN;            // 16384
    if (i < n1) {
        int e = i / DMODEL;
        float v = (e < DINPROJ) ? w_in[(size_t)e * DMODEL + (i % DMODEL)] : 0.f;
        w1h[i] = (_Float16)v;
    } else if (i < n1 + n2) {
        int j = i - n1;
        w2h[j] = (_Float16)w_out[j];
    }
}

// ---------------- kernel 2: block RMSNorm of u, emit f16 rows --------------
__global__ void k_rmsnorm_u(const float* __restrict__ x,
                            const float* __restrict__ w,      // block_norm_w
                            _Float16* __restrict__ uh) {
    int wave = threadIdx.x >> 5, lane = threadIdx.x & 31;
    int row  = blockIdx.x * 8 + wave;
    const float2 v = *(const float2*)(x + (size_t)row * DMODEL + lane * 2);
    float ss = v.x * v.x + v.y * v.y;
    #pragma unroll
    for (int m = 16; m >= 1; m >>= 1) ss += __shfl_xor(ss, m, 32);
    float scale = rsqrtf(ss * (1.f / DMODEL) + EPS_);
    _Float16* o = uh + (size_t)row * DMODEL + lane * 2;
    o[0] = (_Float16)(v.x * scale * w[lane * 2 + 0]);
    o[1] = (_Float16)(v.y * scale * w[lane * 2 + 1]);
}

// ---------------- kernel 3: in_proj GEMM via WMMA --------------------------
// zx(32768 x 560f32, cols<548 valid) = uh(32768x64) * W1^T
// A fragments held in registers across 5 N-tiles (35 = 5 x 7 groups)
#define G1_NT 5
#define G1_NG 7
__global__ void k_gemm_inproj(const _Float16* __restrict__ uh,
                              const _Float16* __restrict__ w1h,
                              float* __restrict__ zx) {
    int gw   = blockIdx.x * 8 + (threadIdx.x >> 5);  // global wave id
    int lane = threadIdx.x & 31;
    int mt = gw / G1_NG, g = gw % G1_NG;
    int half = lane >> 4, n = lane & 15;
    const _Float16* arow = uh + (size_t)(mt * 16 + n) * DMODEL;
    __builtin_prefetch(arow + 8 * DMODEL, 0, 1);
    v16h a0 = load_a_frag(arow, 0,  half);
    v16h a1 = load_a_frag(arow, 32, half);
    CF c[G1_NT];
    #pragma unroll
    for (int j = 0; j < G1_NT; ++j) c[j].v = v8f{};
    #pragma unroll
    for (int j = 0; j < G1_NT; ++j) {
        int nt = g * G1_NT + j;
        const _Float16* bcol = w1h + (size_t)(nt * 16 + n) * DMODEL;
        v16h b0 = load_b_frag(bcol, 0,  half);
        v16h b1 = load_b_frag(bcol, 32, half);
        c[j].v = wmma_f16(a0, b0, c[j].v);
        c[j].v = wmma_f16(a1, b1, c[j].v);
    }
    #pragma unroll
    for (int j = 0; j < G1_NT; ++j) {
        int col = (g * G1_NT + j) * 16 + n;
        #pragma unroll
        for (int r = 0; r < 8; ++r) {
            int row = mt * 16 + 8 * half + r;
            if (col < DINPROJ) zx[(size_t)row * NPAD1 + col] = c[j].f[r];
        }
    }
}

// ---------------- kernel 4: dt = softplus(raw + dt_bias) -------------------
__global__ void k_dt(const float* __restrict__ zx, const float* __restrict__ dt_bias,
                     float* __restrict__ dt) {
    int i = blockIdx.x * blockDim.x + threadIdx.x;   // ROWS*NHEADS
    int r = i >> 2, h = i & 3;
    float v = zx[(size_t)r * NPAD1 + (DIN + CONVDIM) + h] + dt_bias[h];
    dt[i] = (v > 20.f) ? v : log1pf(__expf(v));
}

// ---------------- kernel 5: causal depthwise conv (K=4) + SiLU -------------
__global__ void k_conv(const float* __restrict__ zx,
                       const float* __restrict__ cw,   // (288,4)
                       const float* __restrict__ cb,   // (288,)
                       float* __restrict__ xconv) {
    int i = blockIdx.x * blockDim.x + threadIdx.x;   // 32*32*288
    int c  = i % CONVDIM;
    int lb = (i / CONVDIM) & 31;
    int b  = i / (CONVDIM * 32);
    float w0 = cw[c*4+0], w1 = cw[c*4+1], w2 = cw[c*4+2], w3 = cw[c*4+3];
    float bias = cb[c];
    const float* src = zx + ((size_t)b * LSEQ) * NPAD1 + DIN + c;
    float* dst = xconv + ((size_t)b * LSEQ) * CONVDIM + c;
    for (int j = 0; j < 32; ++j) {
        int l = lb * 32 + j;
        float acc = bias;
        if (l >= 3) acc += src[(size_t)(l-3) * NPAD1] * w0;
        if (l >= 2) acc += src[(size_t)(l-2) * NPAD1] * w1;
        if (l >= 1) acc += src[(size_t)(l-1) * NPAD1] * w2;
        acc += src[(size_t)l * NPAD1] * w3;
        dst[(size_t)l * CONVDIM] = acc / (1.f + __expf(-acc));   // SiLU
    }
}

// ---------------- kernel 6a: chunked scan, local pass ----------------------
// block per (b,h,chunk): 4096 blocks; thread owns 4 state cells (p, 4n).
// Writes y (local part + D skip), cumulative decays aL, chunk-end local state.
__global__ void k_scan_chunk(const float* __restrict__ xconv,
                             const float* __restrict__ dt,
                             const float* __restrict__ A_log,
                             const float* __restrict__ Dp,
                             float* __restrict__ y,
                             float* __restrict__ Hloc,   // (bh,chunk,64,16)
                             float* __restrict__ aL) {   // (row, h)
    __shared__ float sx[HEADDIM], sB[DSTATE], sC[DSTATE], sdt;
    __shared__ float red[HEADDIM * 4];
    int blk = blockIdx.x;
    int c  = blk & (NCHUNK - 1);
    int bh = blk >> 5;
    int b = bh >> 2, h = bh & 3;
    int t = threadIdx.x, p = t & 63, nq = t >> 6;
    float Ah = -__expf(A_log[h]);
    float Dh = Dp[h];
    float st0 = 0.f, st1 = 0.f, st2 = 0.f, st3 = 0.f;
    float aRun = 1.f;
    for (int j = 0; j < CHS; ++j) {
        int l = c * CHS + j;
        size_t row = (size_t)b * LSEQ + l;
        const float* xr = xconv + row * CONVDIM;
        if (t < 64)       sx[t]      = xr[h * HEADDIM + t];
        else if (t < 80)  sB[t - 64] = xr[DIN + (t - 64)];
        else if (t < 96)  sC[t - 80] = xr[DIN + DSTATE + (t - 80)];
        else if (t == 96) sdt        = dt[row * NHEADS + h];
        __syncthreads();
        float dtv = sdt;
        float dA  = __expf(dtv * Ah);
        aRun *= dA;
        float xp  = sx[p];
        float dx  = dtv * xp;
        int nb = nq * 4;
        st0 = st0 * dA + dx * sB[nb+0];
        st1 = st1 * dA + dx * sB[nb+1];
        st2 = st2 * dA + dx * sB[nb+2];
        st3 = st3 * dA + dx * sB[nb+3];
        red[p * 4 + nq] = st0*sC[nb+0] + st1*sC[nb+1] + st2*sC[nb+2] + st3*sC[nb+3];
        __syncthreads();
        if (t == 0) aL[row * NHEADS + h] = aRun;
        if (nq == 0) {
            float yv = red[p*4] + red[p*4+1] + red[p*4+2] + red[p*4+3] + Dh * xp;
            y[row * DIN + h * HEADDIM + p] = yv;
        }
        __syncthreads();
    }
    size_t base = ((size_t)bh * NCHUNK + c) * 1024 + (size_t)p * DSTATE + nq * 4;
    Hloc[base+0] = st0; Hloc[base+1] = st1; Hloc[base+2] = st2; Hloc[base+3] = st3;
}

// ---------------- kernel 6b: carry propagation (element-wise) --------------
// thread per state cell (bh,p,n): 131072 threads, 32-iter loop, no barriers.
__global__ void k_scan_carry(const float* __restrict__ Hloc,
                             const float* __restrict__ aL,
                             float* __restrict__ hstart) {
    int gid = blockIdx.x * blockDim.x + threadIdx.x;
    int bh = gid >> 10, cell = gid & 1023;
    int b = bh >> 2, h = bh & 3;
    float hc = 0.f;
    for (int c = 0; c < NCHUNK; ++c) {
        size_t idx = ((size_t)bh * NCHUNK + c) * 1024 + cell;
        hstart[idx] = hc;
        float Ac = aL[((size_t)b * LSEQ + c * CHS + (CHS-1)) * NHEADS + h];
        hc = Ac * hc + Hloc[idx];
    }
}

// ---------------- kernel 6c: carry fix-up ----------------------------------
// block per row; thread per (h,p):  y += a_l * (C_l . h_start_chunk[p,:])
__global__ void k_scan_fix(const float* __restrict__ hstart,
                           const float* __restrict__ aL,
                           const float* __restrict__ xconv,
                           float* __restrict__ y) {
    __shared__ float sC[DSTATE];
    int row = blockIdx.x;
    int t = threadIdx.x, h = t >> 6, p = t & 63;
    int b = row >> 10, l = row & (LSEQ - 1), c = l >> 5;   // chunk index
    if (t < DSTATE) sC[t] = xconv[(size_t)row * CONVDIM + DIN + DSTATE + t];
    __syncthreads();
    int bh = b * NHEADS + h;
    const float* hs = hstart + ((size_t)bh * NCHUNK + c) * 1024 + (size_t)p * DSTATE;
    float dot = 0.f;
    #pragma unroll
    for (int n = 0; n < DSTATE; ++n) dot += hs[n] * sC[n];
    float a = aL[(size_t)row * NHEADS + h];
    y[(size_t)row * DIN + h * HEADDIM + p] += a * dot;
}

// ---------------- kernel 7: gated RMSNorm -> f16 ---------------------------
__global__ void k_gatenorm(const float* __restrict__ y,
                           const float* __restrict__ zx,
                           const float* __restrict__ ngw,
                           _Float16* __restrict__ yh) {
    int wave = threadIdx.x >> 5, lane = threadIdx.x & 31;
    int row  = blockIdx.x * 8 + wave;
    const float* yr = y  + (size_t)row * DIN   + lane * 8;
    const float* zr = zx + (size_t)row * NPAD1 + lane * 8;   // z = cols [0,256)
    float v[8]; float ss = 0.f;
    #pragma unroll
    for (int j = 0; j < 8; ++j) {
        float zz = zr[j];
        float g  = yr[j] * (zz / (1.f + __expf(-zz)));
        v[j] = g; ss += g * g;
    }
    #pragma unroll
    for (int m = 16; m >= 1; m >>= 1) ss += __shfl_xor(ss, m, 32);
    float scale = rsqrtf(ss * (1.f / DIN) + EPS_);
    _Float16* o = yh + (size_t)row * DIN + lane * 8;
    #pragma unroll
    for (int j = 0; j < 8; ++j) o[j] = (_Float16)(v[j] * scale * ngw[lane * 8 + j]);
}

// ---------------- kernel 8: out_proj GEMM via WMMA + fused residual --------
// 2 N-tiles per wave; A fragment reused across both tiles each K-step.
__global__ void k_gemm_outproj(const _Float16* __restrict__ yh,
                               const _Float16* __restrict__ w2h,
                               const float* __restrict__ x,
                               float* __restrict__ out) {
    int gw   = blockIdx.x * 8 + (threadIdx.x >> 5);
    int lane = threadIdx.x & 31;
    int mt = gw >> 1, g = gw & 1;
    int half = lane >> 4, n = lane & 15;
    const _Float16* arow = yh  + (size_t)(mt * 16 + n) * DIN;
    const _Float16* b0c  = w2h + (size_t)((g * 2 + 0) * 16 + n) * DIN;
    const _Float16* b1c  = w2h + (size_t)((g * 2 + 1) * 16 + n) * DIN;
    __builtin_prefetch(arow + 16 * DIN, 0, 1);
    CF c0, c1; c0.v = v8f{}; c1.v = v8f{};
    #pragma unroll
    for (int kk = 0; kk < DIN; kk += 32) {
        v16h a = load_a_frag(arow, kk, half);
        c0.v = wmma_f16(a, load_b_frag(b0c, kk, half), c0.v);
        c1.v = wmma_f16(a, load_b_frag(b1c, kk, half), c1.v);
    }
    #pragma unroll
    for (int r = 0; r < 8; ++r) {
        int row = mt * 16 + 8 * half + r;
        size_t i0 = (size_t)row * DMODEL + (g * 2 + 0) * 16 + n;  // == b*65536+l*64+col
        size_t i1 = (size_t)row * DMODEL + (g * 2 + 1) * 16 + n;
        out[i0] = c0.f[r] + 2.f * x[i0];          // hidden + batch + x
        out[i1] = c1.f[r] + 2.f * x[i1];
    }
}

// ---------------- host-side launcher ---------------------------------------
extern "C" void kernel_launch(void* const* d_in, const int* in_sizes, int n_in,
                              void* d_out, int out_size, void* d_ws, size_t ws_size,
                              hipStream_t stream) {
    (void)in_sizes; (void)n_in; (void)out_size; (void)ws_size;
    const float* x       = (const float*)d_in[0];
    const float* w_in    = (const float*)d_in[1];
    const float* conv_w  = (const float*)d_in[2];
    const float* conv_b  = (const float*)d_in[3];
    const float* dt_bias = (const float*)d_in[4];
    const float* A_log   = (const float*)d_in[5];
    const float* Dp      = (const float*)d_in[6];
    const float* ngw     = (const float*)d_in[7];
    const float* w_out   = (const float*)d_in[8];
    const float* bnw     = (const float*)d_in[9];
    float* out = (float*)d_out;

    char* ws = (char*)d_ws; size_t off = 0;
    auto alloc = [&](size_t bytes) -> void* {
        void* p = ws + off; off = (off + bytes + 255) & ~(size_t)255; return p;
    };
    _Float16* uh  = (_Float16*)alloc((size_t)ROWS * DMODEL * 2);
    _Float16* w1h = (_Float16*)alloc((size_t)NPAD1 * DMODEL * 2);
    _Float16* w2h = (_Float16*)alloc((size_t)DMODEL * DIN * 2);
    float* zx     = (float*)alloc((size_t)ROWS * NPAD1 * 4);
    float* xconv  = (float*)alloc((size_t)ROWS * CONVDIM * 4);
    float* dtb    = (float*)alloc((size_t)ROWS * NHEADS * 4);
    float* ybuf   = (float*)alloc((size_t)ROWS * DIN * 4);
    _Float16* yh  = (_Float16*)alloc((size_t)ROWS * DIN * 2);
    float* Hloc   = (float*)alloc((size_t)B_ * NHEADS * NCHUNK * 1024 * 4);
    float* hstart = (float*)alloc((size_t)B_ * NHEADS * NCHUNK * 1024 * 4);
    float* aLb    = (float*)alloc((size_t)ROWS * NHEADS * 4);

    // 1) weights -> f16 (pad in_proj N to 560)
    {
        int n = NPAD1 * DMODEL + DMODEL * DIN;
        k_prep_weights<<<(n + 255) / 256, 256, 0, stream>>>(w_in, w_out, w1h, w2h);
    }
    // 2) block RMSNorm -> f16
    k_rmsnorm_u<<<ROWS / 8, 256, 0, stream>>>(x, bnw, uh);
    // 3) in_proj GEMM (WMMA): 2048 M-tiles x 7 groups of 5 N-tiles
    k_gemm_inproj<<<(ROWS / 16) * G1_NG / 8, 256, 0, stream>>>(uh, w1h, zx);
    // 4) dt softplus
    k_dt<<<(ROWS * NHEADS) / 256, 256, 0, stream>>>(zx, dt_bias, dtb);
    // 5) causal conv + SiLU
    k_conv<<<(B_ * 32 * CONVDIM) / 256, 256, 0, stream>>>(zx, conv_w, conv_b, xconv);
    // 6) chunked SSM scan: local (parallel) -> carry (element-wise) -> fix-up
    k_scan_chunk<<<B_ * NHEADS * NCHUNK, 256, 0, stream>>>(xconv, dtb, A_log, Dp,
                                                           ybuf, Hloc, aLb);
    k_scan_carry<<<(B_ * NHEADS * 1024) / 256, 256, 0, stream>>>(Hloc, aLb, hstart);
    k_scan_fix<<<ROWS, 256, 0, stream>>>(hstart, aLb, xconv, ybuf);
    // 7) gated RMSNorm -> f16
    k_gatenorm<<<ROWS / 8, 256, 0, stream>>>(ybuf, zx, ngw, yh);
    // 8) out_proj GEMM (WMMA, 2 N-tiles/wave) + fused residual (+2x)
    k_gemm_outproj<<<(ROWS / 16) * 2 / 8, 256, 0, stream>>>(yh, w2h, x, out);
}
